// MHCrossAttn_5875515261379
// MI455X (gfx1250) — compile-verified, matrix operand order checked by reading, not run on hardware
//
#include <hip/hip_runtime.h>
#include <hip/hip_bf16.h>

// ---------------------------------------------------------------------------
// MHCrossAttn for MI455X (gfx1250), bf16 WMMA path + async LDS copies.
// Pipeline:
//   0) conv_bf16 x7 : pack f32 -> bf16 (query,key,value,Wq,Wk,Wv,Wo)
//   1) gemm_bf16 x3 : Q/K/V projections (bf16 in -> bf16 [B,H,S,64] out),
//                     tiles staged with GLOBAL_LOAD_ASYNC_TO_LDS_B128
//   2) build_aug_q  : Q-tilde  [B,H,SQ,160]  (folds rk_e/rk_p/cross_pos_* )
//   3) build_aug_k  : K-tilde  [B,H,SKV,160] (folds pos_enc/pos_pad/rq_*  )
//   4) build_vtilde : V-tilde^T [B,H,64,SKV] (folds pos_enc*rv_e+pos_pad*rv_p)
//   5) attn_kernel  : per 16-query stripe: scores in LDS (64KB), softmax, PV
//   6) gemm_bf16    : output projection with Wo -> f32 d_out
// ---------------------------------------------------------------------------

typedef __attribute__((ext_vector_type(16))) __bf16 v16bf;
typedef __attribute__((ext_vector_type(8)))  float  v8f;

union Frag {
    v16bf v;
    uint4 u[2];
    unsigned short s[16];
};

__device__ __forceinline__ unsigned short f2bf(float f) {
    unsigned int u = __builtin_bit_cast(unsigned int, f);
    u += 0x7FFFu + ((u >> 16) & 1u);   // round-to-nearest-even
    return (unsigned short)(u >> 16);
}
__device__ __forceinline__ float bf2f(unsigned short s) {
    unsigned int u = ((unsigned int)s) << 16;
    return __builtin_bit_cast(float, u);
}

// Async 16B copy global -> LDS (CDNA5 VGLOBAL, tracked by ASYNCcnt).
__device__ __forceinline__ void async_copy_b128(const void* gptr, void* lds_ptr) {
    unsigned lds = (unsigned)(unsigned long long)lds_ptr;       // LDS byte offset
    unsigned long long ga = (unsigned long long)gptr;
    asm volatile("global_load_async_to_lds_b128 %0, %1, off"
                 :: "v"(lds), "v"(ga) : "memory");
}
__device__ __forceinline__ void async_wait0() {
    asm volatile("s_wait_asynccnt 0x0" ::: "memory");
}

__device__ __forceinline__ Frag load_frag(const unsigned short* p) {
    Frag f;
    f.u[0] = *(const uint4*)p;
    f.u[1] = *(const uint4*)(p + 16);
    return f;
}

// ---------------------------------------------------------------------------
// Pack f32 -> bf16, 4 elements per thread (b128 load, b64 store).
// ---------------------------------------------------------------------------
__global__ __launch_bounds__(256) void conv_bf16(const float* __restrict__ in,
                                                 unsigned short* __restrict__ out,
                                                 int n4) {
    int i = blockIdx.x * blockDim.x + threadIdx.x;
    if (i >= n4) return;
    float4 f = ((const float4*)in)[i];
    unsigned long long pk = (unsigned long long)f2bf(f.x)
                          | ((unsigned long long)f2bf(f.y) << 16)
                          | ((unsigned long long)f2bf(f.z) << 32)
                          | ((unsigned long long)f2bf(f.w) << 48);
    ((unsigned long long*)out)[i] = pk;
}

// ---------------------------------------------------------------------------
// GEMM: C[M,N] = A(bf16, MxKd row-major) * W(bf16, NxKd row-major)^T
// mode 0: bf16 out, head-split [B=row/1024, H=col/64, S=row%1024, DK=col%64]
// mode 1: f32 out, plain row-major MxN
// block = 128 threads (4 waves), tile 64x64, K-step 64.
// Tiles staged via async global->LDS b128 copies; per K-step each wave
// preloads 10 fragments (20 ds_load_b128) then fires 8 WMMAs back-to-back.
// ---------------------------------------------------------------------------
__global__ __launch_bounds__(128) void gemm_bf16(const unsigned short* __restrict__ A,
                                                 const unsigned short* __restrict__ W,
                                                 void* __restrict__ Cout,
                                                 int M, int N, int Kd, int mode) {
    __shared__ __align__(16) unsigned short sA[64][64];   // 8 KB
    __shared__ __align__(16) unsigned short sB[64][64];   // 8 KB
    const int tid  = threadIdx.x;
    const int wave = tid >> 5;
    const int lane = tid & 31;
    const int m0 = blockIdx.x * 64;
    const int n0 = blockIdx.y * 64;

    v8f acc[4] = {};

    const int row  = lane & 15;
    const int koff = (lane & 16) ? 8 : 0;

    for (int k0 = 0; k0 < Kd; k0 += 64) {
        // 64x64 bf16 tile = 512 16B-chunks; 128 threads -> 4 chunks each (x2 mats)
        #pragma unroll
        for (int t = 0; t < 4; ++t) {
            int c = tid + t * 128;          // chunk id 0..511
            int r = c >> 3, c8 = (c & 7) * 8;
            async_copy_b128(A + (size_t)(m0 + r) * Kd + k0 + c8, &sA[r][c8]);
            async_copy_b128(W + (size_t)(n0 + r) * Kd + k0 + c8, &sB[r][c8]);
        }
        if (k0 + 64 < Kd) {                 // prefetch next K-tile
            int c = tid;
            int r = c >> 3, c8 = (c & 7) * 8;
            __builtin_prefetch(A + (size_t)(m0 + r) * Kd + k0 + 64 + c8, 0, 1);
            __builtin_prefetch(W + (size_t)(n0 + r) * Kd + k0 + 64 + c8, 0, 1);
        }
        async_wait0();
        __syncthreads();

        // preload all fragments for this 64-wide K tile, then burst 8 WMMAs
        Frag bfr[2], afr[2][4];
        #pragma unroll
        for (int kk2 = 0; kk2 < 2; ++kk2) {
            bfr[kk2] = load_frag(&sB[wave * 16 + row][kk2 * 32 + koff]);
            #pragma unroll
            for (int mi = 0; mi < 4; ++mi)
                afr[kk2][mi] = load_frag(&sA[mi * 16 + row][kk2 * 32 + koff]);
        }
        #pragma unroll
        for (int kk2 = 0; kk2 < 2; ++kk2)
            #pragma unroll
            for (int mi = 0; mi < 4; ++mi)
                acc[mi] = __builtin_amdgcn_wmma_f32_16x16x32_bf16(
                    false, afr[kk2][mi].v, false, bfr[kk2].v, (short)0,
                    acc[mi], false, false);
        __syncthreads();
    }

    const int mbase = (lane & 16) ? 8 : 0;
    const int ncol  = n0 + wave * 16 + (lane & 15);
    #pragma unroll
    for (int mi = 0; mi < 4; ++mi) {
        #pragma unroll
        for (int i = 0; i < 8; ++i) {
            int mrow = m0 + mi * 16 + i + mbase;
            float val = acc[mi][i];
            if (mode == 0) {
                int b = mrow >> 10, q = mrow & 1023, h = ncol >> 6, dk = ncol & 63;
                ((unsigned short*)Cout)[((((size_t)b * 16 + h) << 10) + q) * 64 + dk] = f2bf(val);
            } else {
                ((float*)Cout)[(size_t)mrow * N + ncol] = val;
            }
        }
    }
}

// ---------------------------------------------------------------------------
// Q-tilde: [bh,q,0:64]=Q ; [64+r]=Q.rk_e ; [81+r]=Q.rk_p ;
//          [98+r]=cross_pos_enc[r,q] ; [115+r]=cross_pos_pad[r,q] ; [132:160]=0
// ---------------------------------------------------------------------------
__global__ __launch_bounds__(128) void build_aug_q(const unsigned short* __restrict__ Qh,
                                                   const float* __restrict__ relk,
                                                   const float* __restrict__ cpe,
                                                   const float* __restrict__ cpp,
                                                   unsigned short* __restrict__ Qt) {
    int idx = blockIdx.x * blockDim.x + threadIdx.x;   // bh*1024 + q
    if (idx >= 64 * 1024) return;
    int bh = idx >> 10, q = idx & 1023, h = bh & 15;
    const unsigned short* qrow = Qh + (size_t)idx * 64;
    unsigned short* orow = Qt + (size_t)idx * 160;
    float qv[64];
    #pragma unroll
    for (int d = 0; d < 64; ++d) { qv[d] = bf2f(qrow[d]); orow[d] = qrow[d]; }
    for (int r = 0; r < 17; ++r) {
        const float* re = relk + ((size_t)h * 34 + r) * 64;
        const float* rp = relk + ((size_t)h * 34 + 17 + r) * 64;
        float se = 0.f, sp = 0.f;
        #pragma unroll
        for (int d = 0; d < 64; ++d) { se += qv[d] * re[d]; sp += qv[d] * rp[d]; }
        orow[64 + r] = f2bf(se);
        orow[81 + r] = f2bf(sp);
    }
    for (int r = 0; r < 17; ++r) {
        orow[98 + r]  = f2bf(cpe[((size_t)bh * 17 + r) * 1024 + q]);
        orow[115 + r] = f2bf(cpp[((size_t)bh * 17 + r) * 1024 + q]);
    }
    for (int c = 132; c < 160; ++c) orow[c] = 0;
}

// ---------------------------------------------------------------------------
// K-tilde: [bh,s,0:64]=K ; [64+r]=pos_enc[r,s] ; [81+r]=pos_pad[r,s] ;
//          [98+r]=K.rq_e ; [115+r]=K.rq_p ; [132:160]=0
// ---------------------------------------------------------------------------
__global__ __launch_bounds__(128) void build_aug_k(const unsigned short* __restrict__ Kh,
                                                   const float* __restrict__ relq,
                                                   const float* __restrict__ pe,
                                                   const float* __restrict__ pp,
                                                   unsigned short* __restrict__ Kt) {
    int idx = blockIdx.x * blockDim.x + threadIdx.x;   // bh*1024 + s
    if (idx >= 64 * 1024) return;
    int bh = idx >> 10, s = idx & 1023, h = bh & 15;
    const unsigned short* krow = Kh + (size_t)idx * 64;
    unsigned short* orow = Kt + (size_t)idx * 160;
    float kv[64];
    #pragma unroll
    for (int d = 0; d < 64; ++d) { kv[d] = bf2f(krow[d]); orow[d] = krow[d]; }
    for (int r = 0; r < 17; ++r) {
        orow[64 + r] = f2bf(pe[((size_t)bh * 17 + r) * 1024 + s]);
        orow[81 + r] = f2bf(pp[((size_t)bh * 17 + r) * 1024 + s]);
    }
    for (int r = 0; r < 17; ++r) {
        const float* re = relq + ((size_t)h * 34 + r) * 64;
        const float* rp = relq + ((size_t)h * 34 + 17 + r) * 64;
        float se = 0.f, sp = 0.f;
        #pragma unroll
        for (int d = 0; d < 64; ++d) { se += kv[d] * re[d]; sp += kv[d] * rp[d]; }
        orow[98 + r]  = f2bf(se);
        orow[115 + r] = f2bf(sp);
    }
    for (int c = 132; c < 160; ++c) orow[c] = 0;
}

// ---------------------------------------------------------------------------
// V-tilde transposed: Vt[bh][d][s] = V[bh][s][d] + sum_r pe[r,s]*rv_e[r,d]
//                                               + sum_r pp[r,s]*rv_p[r,d]
// ---------------------------------------------------------------------------
__global__ __launch_bounds__(128) void build_vtilde(const unsigned short* __restrict__ Vh,
                                                    const float* __restrict__ relv,
                                                    const float* __restrict__ pe,
                                                    const float* __restrict__ pp,
                                                    unsigned short* __restrict__ Vt) {
    int idx = blockIdx.x * blockDim.x + threadIdx.x;   // bh*1024 + s
    if (idx >= 64 * 1024) return;
    int bh = idx >> 10, s = idx & 1023, h = bh & 15;
    float pev[17], ppv[17];
    for (int r = 0; r < 17; ++r) {
        pev[r] = pe[((size_t)bh * 17 + r) * 1024 + s];
        ppv[r] = pp[((size_t)bh * 17 + r) * 1024 + s];
    }
    const unsigned short* vrow = Vh + (size_t)idx * 64;
    for (int d = 0; d < 64; ++d) {
        float acc = bf2f(vrow[d]);
        for (int r = 0; r < 17; ++r) {
            acc += pev[r] * relv[((size_t)h * 34 + r) * 64 + d];
            acc += ppv[r] * relv[((size_t)h * 34 + 17 + r) * 64 + d];
        }
        Vt[((size_t)bh * 64 + d) * 1024 + s] = f2bf(acc);
    }
}

// ---------------------------------------------------------------------------
// Attention: grid (SQ/16, B*H), 128 threads (4 waves).
// Phase 1: scores(16 x 1024) = Qt(16x160) * Kt^T, f32 in LDS (64 KB),
//          waves split the 64 s-chunks; preload 5 K-frags then 5 WMMAs.
// Phase 2: softmax in LDS, 8-thread row teams with __shfl_xor reductions.
// Phase 3: PV, waves split 32-wide s-chunks; attn f32->bf16 A-frags,
//          preload 4 V-frags then 4 WMMAs -> 16x64 partial O per wave.
// Phase 4: cross-wave reduction through (reused) score LDS, /rowsum,
//          store bf16 into [B,SQ,DM].
// ---------------------------------------------------------------------------
__global__ __launch_bounds__(128) void attn_kernel(const unsigned short* __restrict__ Qt,
                                                   const unsigned short* __restrict__ Kt,
                                                   const unsigned short* __restrict__ Vt,
                                                   unsigned short* __restrict__ Ob) {
    __shared__ __align__(16) float sc[16][1024];   // 64 KB (gfx1250: 320KB/WG cap)
    __shared__ float s_rowsum[16];

    const int bh = blockIdx.y;
    const int b = bh >> 4, h = bh & 15;
    const int q0 = blockIdx.x * 16;
    const int tid = threadIdx.x, wave = tid >> 5, lane = tid & 31;
    const int AK = 160;

    const unsigned short* Qbase = Qt + ((size_t)bh * 1024 + q0) * AK;
    const unsigned short* Kbase = Kt + (size_t)bh * 1024 * AK;
    const unsigned short* Vbase = Vt + (size_t)bh * 64 * 1024;

    const int row   = lane & 15;
    const int koff  = (lane & 16) ? 8 : 0;
    const int mbase = (lane & 16) ? 8 : 0;
    const int ncol  = lane & 15;

    // Q fragments for this 16-row stripe (k = 0..159 in 5 steps of 32)
    Frag qf[5];
    #pragma unroll
    for (int kk = 0; kk < 5; ++kk)
        qf[kk] = load_frag(Qbase + (size_t)row * AK + kk * 32 + koff);

    // ---- Phase 1: scores ----
    for (int c16 = wave; c16 < 64; c16 += 4) {
        int s0 = c16 * 16;
        Frag kf[5];
        #pragma unroll
        for (int kk = 0; kk < 5; ++kk)
            kf[kk] = load_frag(Kbase + (size_t)(s0 + row) * AK + kk * 32 + koff);
        v8f acc = {};
        #pragma unroll
        for (int kk = 0; kk < 5; ++kk)
            acc = __builtin_amdgcn_wmma_f32_16x16x32_bf16(
                false, qf[kk].v, false, kf[kk].v, (short)0, acc, false, false);
        #pragma unroll
        for (int i = 0; i < 8; ++i)
            sc[i + mbase][s0 + ncol] = acc[i] * 0.125f;   // 1/sqrt(64)
    }
    __syncthreads();

    // ---- Phase 2: softmax (unnormalized probs stay in LDS) ----
    {
        int r = tid >> 3;       // row 0..15, 8-thread team (contiguous lanes)
        int c0 = tid & 7;
        float mx = -1e30f;
        for (int s = c0; s < 1024; s += 8) mx = fmaxf(mx, sc[r][s]);
        #pragma unroll
        for (int off = 1; off < 8; off <<= 1) mx = fmaxf(mx, __shfl_xor(mx, off, 32));
        float sum = 0.f;
        for (int s = c0; s < 1024; s += 8) {
            float p = __expf(sc[r][s] - mx);
            sc[r][s] = p;
            sum += p;
        }
        #pragma unroll
        for (int off = 1; off < 8; off <<= 1) sum += __shfl_xor(sum, off, 32);
        if (c0 == 0) s_rowsum[r] = sum;
    }
    __syncthreads();

    // ---- Phase 3: PV ----
    v8f oacc[4] = {};
    for (int c32 = wave; c32 < 32; c32 += 4) {
        int s0 = c32 * 32;
        Frag vf[4];
        #pragma unroll
        for (int ni = 0; ni < 4; ++ni)     // Vt^T rows are d-columns: contiguous k
            vf[ni] = load_frag(Vbase + (size_t)(ni * 16 + row) * 1024 + s0 + koff);
        Frag af;                           // attn 16x32 A-fragment (f32->bf16)
        #pragma unroll
        for (int j = 0; j < 16; ++j) {
            int k = koff + (j & 7) + ((j >> 3) << 4);
            af.s[j] = f2bf(sc[row][s0 + k]);
        }
        #pragma unroll
        for (int ni = 0; ni < 4; ++ni)
            oacc[ni] = __builtin_amdgcn_wmma_f32_16x16x32_bf16(
                false, af.v, false, vf[ni].v, (short)0, oacc[ni], false, false);
    }
    __syncthreads();                       // scores LDS now dead -> reuse

    // ---- Phase 4: cross-wave reduce, normalize, store bf16 ----
    float* part = (float*)sc;              // [wave][16][64] = 16 KB
    #pragma unroll
    for (int ni = 0; ni < 4; ++ni)
        #pragma unroll
        for (int i = 0; i < 8; ++i)
            part[((size_t)wave * 16 + i + mbase) * 64 + ni * 16 + ncol] = oacc[ni][i];
    __syncthreads();

    for (int idx = tid; idx < 16 * 64; idx += 128) {
        int m = idx >> 6, d = idx & 63;
        float v = part[m * 64 + d] + part[(16 + m) * 64 + d] +
                  part[(32 + m) * 64 + d] + part[(48 + m) * 64 + d];
        v /= s_rowsum[m];
        Ob[(((size_t)b * 1024) + q0 + m) * 1024 + (size_t)h * 64 + d] = f2bf(v);
    }
}

// ---------------------------------------------------------------------------
extern "C" void kernel_launch(void* const* d_in, const int* in_sizes, int n_in,
                              void* d_out, int out_size, void* d_ws, size_t ws_size,
                              hipStream_t stream) {
    const float* query = (const float*)d_in[0];
    const float* key   = (const float*)d_in[1];
    const float* value = (const float*)d_in[2];
    const float* pe    = (const float*)d_in[3];   // pos_enc       [B,H,17,SKV]
    const float* pp    = (const float*)d_in[4];   // pos_pad
    const float* cpe   = (const float*)d_in[5];   // cross_pos_enc [B,H,17,SQ]
    const float* cpp   = (const float*)d_in[6];   // cross_pos_pad
    const float* rel_q = (const float*)d_in[7];   // [1,H,34,64]
    const float* rel_k = (const float*)d_in[8];
    const float* rel_v = (const float*)d_in[9];
    const float* Wq    = (const float*)d_in[10];
    const float* Wk    = (const float*)d_in[11];
    const float* Wv    = (const float*)d_in[12];
    const float* Wo    = (const float*)d_in[13];
    float* out = (float*)d_out;

    char* ws = (char*)d_ws;
    const size_t MB = 1ull << 20;
    unsigned short* Xq  = (unsigned short*)(ws);              //  8 MB [B,S,DM] bf16
    unsigned short* Xk  = (unsigned short*)(ws + 8   * MB);   //  8 MB
    unsigned short* Xv  = (unsigned short*)(ws + 16  * MB);   //  8 MB
    unsigned short* Wqb = (unsigned short*)(ws + 24  * MB);   //  2 MB [DM,DM] bf16
    unsigned short* Wkb = (unsigned short*)(ws + 26  * MB);   //  2 MB
    unsigned short* Wvb = (unsigned short*)(ws + 28  * MB);   //  2 MB
    unsigned short* Wob = (unsigned short*)(ws + 30  * MB);   //  2 MB
    unsigned short* Qh  = (unsigned short*)(ws + 32  * MB);   //  8 MB [B,H,1024,64]
    unsigned short* Kh  = (unsigned short*)(ws + 40  * MB);   //  8 MB
    unsigned short* Vh  = (unsigned short*)(ws + 48  * MB);   //  8 MB
    unsigned short* Qt  = (unsigned short*)(ws + 56  * MB);   // 20 MB [B,H,1024,160]
    unsigned short* Kt  = (unsigned short*)(ws + 76  * MB);   // 20 MB
    unsigned short* Vt  = (unsigned short*)(ws + 96  * MB);   //  8 MB [B,H,64,1024]
    unsigned short* Obf = (unsigned short*)(ws + 104 * MB);   //  8 MB [B,S,DM] bf16

    // 0) pack everything the GEMMs touch to bf16
    int nx4 = 4096 * 1024 / 4, nw4 = 1024 * 1024 / 4;
    conv_bf16<<<(nx4 + 255) / 256, 256, 0, stream>>>(query, Xq, nx4);
    conv_bf16<<<(nx4 + 255) / 256, 256, 0, stream>>>(key,   Xk, nx4);
    conv_bf16<<<(nx4 + 255) / 256, 256, 0, stream>>>(value, Xv, nx4);
    conv_bf16<<<(nw4 + 255) / 256, 256, 0, stream>>>(Wq, Wqb, nw4);
    conv_bf16<<<(nw4 + 255) / 256, 256, 0, stream>>>(Wk, Wkb, nw4);
    conv_bf16<<<(nw4 + 255) / 256, 256, 0, stream>>>(Wv, Wvb, nw4);
    conv_bf16<<<(nw4 + 255) / 256, 256, 0, stream>>>(Wo, Wob, nw4);

    dim3 ggrid(4096 / 64, 1024 / 64);   // M tiles x N tiles
    gemm_bf16<<<ggrid, 128, 0, stream>>>(Xq, Wqb, Qh, 4096, 1024, 1024, 0);
    gemm_bf16<<<ggrid, 128, 0, stream>>>(Xk, Wkb, Kh, 4096, 1024, 1024, 0);
    gemm_bf16<<<ggrid, 128, 0, stream>>>(Xv, Wvb, Vh, 4096, 1024, 1024, 0);

    int n1 = 64 * 1024;                 // B*H*S
    build_aug_q<<<n1 / 128, 128, 0, stream>>>(Qh, rel_k, cpe, cpp, Qt);
    build_aug_k<<<n1 / 128, 128, 0, stream>>>(Kh, rel_q, pe, pp, Kt);
    build_vtilde<<<n1 / 128, 128, 0, stream>>>(Vh, rel_v, pe, pp, Vt);

    attn_kernel<<<dim3(64, 64), 128, 0, stream>>>(Qt, Kt, Vt, Obf);

    gemm_bf16<<<ggrid, 128, 0, stream>>>(Obf, Wob, out, 4096, 1024, 1024, 1);

    (void)in_sizes; (void)n_in; (void)out_size; (void)ws_size;
}
